// DiffusionModel_66932770341075
// MI455X (gfx1250) — compile-verified, matrix operand order checked by reading, not run on hardware
//
#include <hip/hip_runtime.h>
#include <hip/hip_bf16.h>

// ---------------------------------------------------------------------------
// CDNA5 (gfx1250) persistent-kernel reverse diffusion.
//   - 8 workgroups x 512 threads (16 wave32s). WG b owns batch rows [16b,16b+16):
//     the recurrence is independent per batch row -> no inter-WG sync ever.
//   - GEMMs via v_wmma_f32_16x16x32_f16 (f16 inputs, f32 accum).
//   - Weights pre-transposed+cast to f16 [N][Kpad]: every WMMA fragment is two
//     contiguous 16B loads per lane (global b128 for B, ds_load_b128 for A).
//   - Recurrent state x kept in f32 registers (16 values/thread) for 1000 steps.
//   - K-loop kept as a real loop (runtime trip count + unroll 2) so the
//     compiler cannot fully unroll and spill B fragments to scratch.
// ---------------------------------------------------------------------------

typedef _Float16 v8h_t  __attribute__((ext_vector_type(8)));
typedef _Float16 v16h_t __attribute__((ext_vector_type(16)));
typedef float    v8f_t  __attribute__((ext_vector_type(8)));

static constexpr int T_STEPS = 1000;
static constexpr int BATCH   = 128;
static constexpr int DIM     = 512;
static constexpr int K1      = 513;   // D+1 (x concat t_emb)
static constexpr int K1P     = 544;   // 513 padded to multiple of 32 (17*32)
static constexpr int LDA1    = 544;   // stride of act buffer A (f16 elems)
static constexpr int LDB     = 512;   // stride of act buffer B (f16 elems)

// workspace layout (bytes)
static constexpr size_t OFF_W1T = 0;                      // 512*544*2 = 557056
static constexpr size_t OFF_W2T = 557056;                 // 512*512*2 = 524288
static constexpr size_t OFF_W3T = OFF_W2T + 524288;
static constexpr size_t OFF_W4T = OFF_W3T + 524288;
static constexpr size_t OFF_TAB = OFF_W4T + 524288;       // 1000 * float4

// ---------------------------------------------------------------------------
// Prep kernel 1: cast+transpose one weight matrix f32[K][N] -> f16[N][LDK],
// zero-padding K..LDK-1 (layer-1 pads 513 -> 544).
// ---------------------------------------------------------------------------
__global__ void transpose_cast(const float* __restrict__ src,
                               _Float16* __restrict__ dst,
                               int K, int N, int LDK) {
  int i = blockIdx.x * blockDim.x + threadIdx.x;
  int total = N * LDK;
  if (i >= total) return;
  int n = i / LDK, k = i - n * LDK;
  dst[(size_t)n * LDK + k] =
      (k < K) ? (_Float16)src[(size_t)k * N + n] : (_Float16)0.f;
}

// ---------------------------------------------------------------------------
// Prep kernel 2: per-timestep scalar table.
//   table[t] = (A_t, B_t, sigma_t, t_emb_t) where
//   x_next = A_t * x + B_t * noise_pred + sigma_t * noise.
//   t_emb is a scalar: the time-MLP input (t/T) is batch-uniform.
// ---------------------------------------------------------------------------
__global__ void prep_table(const float* __restrict__ tw1,
                           const float* __restrict__ tb1,
                           const float* __restrict__ tw2,
                           const float* __restrict__ tb2,
                           float4* __restrict__ table) {
  int t = blockIdx.x * blockDim.x + threadIdx.x;
  if (t >= T_STEPS) return;
  const float bs = 0.0001f, be = 0.02f;
  float ac = 1.f, ac_prev = 1.f, beta_t = 0.f;
  for (int j = 0; j <= t; ++j) {                 // cumprod(alphas)[0..t]
    float beta = bs + (be - bs) * (float)j / (float)(T_STEPS - 1);
    if (j == t) { ac_prev = ac; beta_t = beta; } // ac_prev == ac[t-1] (==1 at t=0)
    ac *= (1.f - beta);
  }
  float sat = sqrtf(ac);
  float At = sqrtf(ac_prev) / sat;
  float Bt = sqrtf(1.f - ac_prev) - sqrtf(ac_prev) * sqrtf(1.f - ac) / sat;
  float St = (t > 0) ? sqrtf(beta_t) : 0.f;
  // scalar time embedding
  float tin = (float)t / (float)T_STEPS;
  float acc = tb2[0];
  for (int j = 0; j < 128; ++j) {
    float h = tin * tw1[j] + tb1[j];
    h = h > 0.f ? h : 0.f;
    acc += h * tw2[j];
  }
  table[t] = make_float4(At, Bt, St, acc);
}

// ---------------------------------------------------------------------------
// One wave computes two 16x16 output tiles (columns nbase..nbase+31) of
// D = A(16xK) * W(KxN).  A rows live in LDS f16 (row-major, stride lda);
// W is pre-transposed f16 [N][ldk], so both fragments are contiguous loads:
//   lane: half = lane>>4, idx = lane&15
//   A frag elems 0..7  = A[idx][kb*32 + half*8 + 0..7]
//   A frag elems 8..15 = A[idx][kb*32 + 16 + half*8 + 0..7]
//   B frag identical pattern with row 'n' of Wt (i.e. column n of W).
// kIters is laundered through an asm barrier so LLVM cannot fully unroll the
// loop after inlining (full unroll caused scratch spills of B fragments).
// ---------------------------------------------------------------------------
__device__ __forceinline__ void wave_gemm2(const _Float16* __restrict__ aLDS,
                                           int lda,
                                           const _Float16* __restrict__ wt,
                                           int ldk, int kIters, int idx,
                                           int half, int nbase, v8f_t& c0,
                                           v8f_t& c1) {
  asm volatile("" : "+r"(kIters));  // opaque trip count: keep this a real loop
  const _Float16* ap  = aLDS + idx * lda + half * 8;
  const _Float16* bp0 = wt + (size_t)(nbase + idx) * ldk + half * 8;
  const _Float16* bp1 = bp0 + (size_t)16 * ldk;
#pragma unroll 2
  for (int kb = 0; kb < kIters; ++kb) {
    const int ko = kb * 32;
    v8h_t alo = *(const v8h_t*)(ap + ko);
    v8h_t ahi = *(const v8h_t*)(ap + ko + 16);
    v16h_t a = __builtin_shufflevector(alo, ahi, 0, 1, 2, 3, 4, 5, 6, 7, 8, 9,
                                       10, 11, 12, 13, 14, 15);
    v8h_t b0l = *(const v8h_t*)(bp0 + ko);
    v8h_t b0h = *(const v8h_t*)(bp0 + ko + 16);
    v16h_t b0 = __builtin_shufflevector(b0l, b0h, 0, 1, 2, 3, 4, 5, 6, 7, 8, 9,
                                        10, 11, 12, 13, 14, 15);
    c0 = __builtin_amdgcn_wmma_f32_16x16x32_f16(false, a, false, b0, (short)0,
                                                c0, false, false);
    v8h_t b1l = *(const v8h_t*)(bp1 + ko);
    v8h_t b1h = *(const v8h_t*)(bp1 + ko + 16);
    v16h_t b1 = __builtin_shufflevector(b1l, b1h, 0, 1, 2, 3, 4, 5, 6, 7, 8, 9,
                                        10, 11, 12, 13, 14, 15);
    c1 = __builtin_amdgcn_wmma_f32_16x16x32_f16(false, a, false, b1, (short)0,
                                                c1, false, false);
  }
}

// C-tile layout: vgpr j of lane (half,idx) holds D[j + 8*half][idx] of the tile.
__device__ __forceinline__ void store_relu(const v8f_t& c,
                                           const float* __restrict__ bias,
                                           _Float16* __restrict__ dst, int ldd,
                                           int col, int half) {
  float bv = bias[col];
#pragma unroll
  for (int j = 0; j < 8; ++j) {
    float v = c[j] + bv;
    v = v > 0.f ? v : 0.f;
    dst[(j + half * 8) * ldd + col] = (_Float16)v;
  }
}

// ---------------------------------------------------------------------------
// Main persistent kernel: 8 blocks x 512 threads; block b owns rows 16b..16b+15
// and iterates all 1000 steps locally (only workgroup barriers needed).
// 16 waves/WGP = 4 waves/SIMD: tell the allocator so it gives each wave a
// quarter of the VGPR file instead of spilling.
// ---------------------------------------------------------------------------
__global__ __launch_bounds__(512)
__attribute__((amdgpu_waves_per_eu(4))) void diffusion_main(
    const float* __restrict__ x_init, const float* __restrict__ noise,
    const float* __restrict__ b1v, const float* __restrict__ b2v,
    const float* __restrict__ b3v, const float* __restrict__ b4v,
    const _Float16* __restrict__ w1t, const _Float16* __restrict__ w2t,
    const _Float16* __restrict__ w3t, const _Float16* __restrict__ w4t,
    const float4* __restrict__ table, float* __restrict__ out) {
  __shared__ __align__(16) _Float16 actA[16 * LDA1];  // layer-1 input (x|temb|pad)
  __shared__ __align__(16) _Float16 actB[16 * LDB];   // ping-pong activations

  const int tid = threadIdx.x;
  const int wave = tid >> 5;
  const int lane = tid & 31;
  const int half = lane >> 4;
  const int idx = lane & 15;
  const int nbase = wave * 32;      // this wave's output-column base
  const int r0 = blockIdx.x * 16;   // this WG's batch-row base

  // Recurrent state x in f32 registers, same (row,col) mapping as C tiles.
  float xreg[16];
#pragma unroll
  for (int s = 0; s < 2; ++s) {
    int col = nbase + s * 16 + idx;
#pragma unroll
    for (int j = 0; j < 8; ++j) {
      int row = j + half * 8;
      float v = x_init[(size_t)(r0 + row) * DIM + col];
      xreg[s * 8 + j] = v;
      actA[row * LDA1 + col] = (_Float16)v;
    }
  }
  // zero the K-padding columns 512..543 once (col 512 rewritten with temb each step)
  for (int i = tid; i < 16 * 32; i += 512) {
    int m = i >> 5, k = 512 + (i & 31);
    actA[m * LDA1 + k] = (_Float16)0.f;
  }
  __syncthreads();

  for (int it = 0; it < T_STEPS; ++it) {
    const int t = T_STEPS - 1 - it;
    const float4 sc = table[t];  // (A_t, B_t, sigma_t, temb_t)
    if (tid < 16) actA[tid * LDA1 + 512] = (_Float16)sc.w;  // temb column
    __syncthreads();

    v8f_t c0 = {}, c1 = {};
    wave_gemm2(actA, LDA1, w1t, K1P, 17, idx, half, nbase, c0, c1);
    store_relu(c0, b1v, actB, LDB, nbase + idx, half);
    store_relu(c1, b1v, actB, LDB, nbase + 16 + idx, half);
    __syncthreads();

    c0 = {}; c1 = {};
    wave_gemm2(actB, LDB, w2t, DIM, 16, idx, half, nbase, c0, c1);
    store_relu(c0, b2v, actA, LDA1, nbase + idx, half);
    store_relu(c1, b2v, actA, LDA1, nbase + 16 + idx, half);
    __syncthreads();

    c0 = {}; c1 = {};
    wave_gemm2(actA, LDA1, w3t, DIM, 16, idx, half, nbase, c0, c1);
    store_relu(c0, b3v, actB, LDB, nbase + idx, half);
    store_relu(c1, b3v, actB, LDB, nbase + 16 + idx, half);
    __syncthreads();

    c0 = {}; c1 = {};
    wave_gemm2(actB, LDB, w4t, DIM, 16, idx, half, nbase, c0, c1);
    // fused reverse-diffusion update; write next-step layer-1 input directly
#pragma unroll
    for (int s = 0; s < 2; ++s) {
      v8f_t c = s ? c1 : c0;
      int col = nbase + s * 16 + idx;
      float bv = b4v[col];
#pragma unroll
      for (int j = 0; j < 8; ++j) {
        int row = j + half * 8;
        float np = c[j] + bv;
        float nz = noise[((size_t)t * BATCH + (r0 + row)) * DIM + col];
        float xn = sc.x * xreg[s * 8 + j] + sc.y * np + sc.z * nz;
        xreg[s * 8 + j] = xn;
        actA[row * LDA1 + col] = (_Float16)xn;
      }
    }
    // no barrier needed here: next iteration's post-temb barrier orders actA
  }

#pragma unroll
  for (int s = 0; s < 2; ++s) {
    int col = nbase + s * 16 + idx;
#pragma unroll
    for (int j = 0; j < 8; ++j)
      out[(size_t)(r0 + j + half * 8) * DIM + col] = xreg[s * 8 + j];
  }
}

// ---------------------------------------------------------------------------
extern "C" void kernel_launch(void* const* d_in, const int* in_sizes, int n_in,
                              void* d_out, int out_size, void* d_ws,
                              size_t ws_size, hipStream_t stream) {
  (void)in_sizes; (void)n_in; (void)out_size; (void)ws_size;
  const float* x_init = (const float*)d_in[0];
  const float* noise  = (const float*)d_in[1];
  const float* w1 = (const float*)d_in[2];
  const float* b1 = (const float*)d_in[3];
  const float* w2 = (const float*)d_in[4];
  const float* b2 = (const float*)d_in[5];
  const float* w3 = (const float*)d_in[6];
  const float* b3 = (const float*)d_in[7];
  const float* w4 = (const float*)d_in[8];
  const float* b4 = (const float*)d_in[9];
  const float* tw1 = (const float*)d_in[10];
  const float* tb1 = (const float*)d_in[11];
  const float* tw2 = (const float*)d_in[12];
  const float* tb2 = (const float*)d_in[13];

  char* ws = (char*)d_ws;
  _Float16* w1t = (_Float16*)(ws + OFF_W1T);
  _Float16* w2t = (_Float16*)(ws + OFF_W2T);
  _Float16* w3t = (_Float16*)(ws + OFF_W3T);
  _Float16* w4t = (_Float16*)(ws + OFF_W4T);
  float4* table = (float4*)(ws + OFF_TAB);

  int n1 = DIM * K1P;
  transpose_cast<<<(n1 + 255) / 256, 256, 0, stream>>>(w1, w1t, K1, DIM, K1P);
  int n2 = DIM * DIM;
  transpose_cast<<<(n2 + 255) / 256, 256, 0, stream>>>(w2, w2t, DIM, DIM, DIM);
  transpose_cast<<<(n2 + 255) / 256, 256, 0, stream>>>(w3, w3t, DIM, DIM, DIM);
  transpose_cast<<<(n2 + 255) / 256, 256, 0, stream>>>(w4, w4t, DIM, DIM, DIM);
  prep_table<<<4, 256, 0, stream>>>(tw1, tb1, tw2, tb2, table);

  diffusion_main<<<BATCH / 16, 512, 0, stream>>>(
      x_init, noise, b1, b2, b3, b4, w1t, w2t, w3t, w4t, table, (float*)d_out);
}
